// QueryGatedSelfAttention_63127429317121
// MI455X (gfx1250) — compile-verified, matrix operand order checked by reading, not run on hardware
//
#include <hip/hip_runtime.h>

#define B_   8
#define L_   1024
#define D_   512
#define NQ_  16

static __device__ __constant__ float INV_T = 0.044194173824159216f; // 1/sqrt(512)

typedef __attribute__((ext_vector_type(16))) _Float16 v16h;
typedef __attribute__((ext_vector_type(8)))  float    v8f;

__device__ __forceinline__ v8f wmma16(v16h a, v16h b, v8f c) {
    // D = A(16x32 f16) * B(32x16 f16) + C(16x16 f32)
    return __builtin_amdgcn_wmma_f32_16x16x32_f16(false, a, false, b, (short)0, c, false, false);
}

// A-fragment scatter mapping: element (row m, k-within-step wi) -> (lane, pos)
__device__ __forceinline__ void afrag_map(int wi, int& laneOfs, int& pos) {
    if (wi < 8)       { laneOfs = 0;  pos = wi;      }
    else if (wi < 16) { laneOfs = 16; pos = wi - 8;  }
    else if (wi < 24) { laneOfs = 0;  pos = wi - 8;  }
    else              { laneOfs = 16; pos = wi - 16; }
}

// ---------------------------------------------------------------------------
// Kernel 1: one block per (b, 16-row tile).
//   pass1: logits = q k^T / T   (WMMA f16, f32 acc)
//   softmax across 1024 cols (cross-wave LDS reduction)
//   writes attn, log_attn; stages attn (f16) in LDS fragment layout
//   pass2: output = attn @ v -> stashed into outBig[b, 0, l, d]
// ---------------------------------------------------------------------------
__global__ __launch_bounds__(256)
void qgsa_self_attn(const float* __restrict__ q, const float* __restrict__ k,
                    const float* __restrict__ v, float* __restrict__ outBig,
                    float* __restrict__ attnOut, float* __restrict__ logAttnOut)
{
    __shared__ __align__(32) _Float16 frag[16384];   // 32KB: q-frags (16 steps) then attn-frags (32 steps)
    __shared__ float red[8][16][16];                  // 8KB cross-wave reduction
    __shared__ float rowmax[16];
    __shared__ float rowsum[16];

    const int tid    = threadIdx.x;
    const int w      = tid >> 5;        // wave 0..7
    const int ll     = tid & 31;        // lane 0..31
    const int lane15 = ll & 15;
    const int hib    = (ll >= 16) ? 1 : 0;
    const int b      = blockIdx.x >> 6;       // L_/16 = 64 row-tiles per batch
    const int m0     = (blockIdx.x & 63) << 4;

    // ---- stage q tile [16 x 512] into A-fragment layout (f32 -> f16) ----
    for (int e = tid; e < 16 * D_; e += 256) {
        int m = e >> 9;            // row in tile
        int d = e & (D_ - 1);
        float val = q[((size_t)b * L_ + m0 + m) * D_ + d];
        int s = d >> 5, wi = d & 31, lo, pos;
        afrag_map(wi, lo, pos);
        frag[(((s << 5) + m + lo) << 4) + pos] = (_Float16)val;
    }
    __syncthreads();

    // ---- pass 1: logits. wave w owns n-tiles [w*8 .. w*8+7] (cols w*128..w*128+127)
    v8f cacc[8];
    const float* kb = k + (size_t)b * L_ * D_;
    #pragma unroll
    for (int t = 0; t < 8; ++t) {
        v8f c = {0.f, 0.f, 0.f, 0.f, 0.f, 0.f, 0.f, 0.f};
        const float* krow = kb + (size_t)(((w << 3) + t) * 16 + lane15) * D_;
        for (int s = 0; s < 16; ++s) {
            v16h a = *(const v16h*)&frag[((s << 5) + ll) << 4];
            int c0 = (s << 5) + (hib ? 8 : 0);
            float4 x0 = *(const float4*)(krow + c0);
            float4 x1 = *(const float4*)(krow + c0 + 4);
            float4 x2 = *(const float4*)(krow + c0 + 16);
            float4 x3 = *(const float4*)(krow + c0 + 20);
            v16h bv;
            bv[0]=(_Float16)x0.x;  bv[1]=(_Float16)x0.y;  bv[2]=(_Float16)x0.z;  bv[3]=(_Float16)x0.w;
            bv[4]=(_Float16)x1.x;  bv[5]=(_Float16)x1.y;  bv[6]=(_Float16)x1.z;  bv[7]=(_Float16)x1.w;
            bv[8]=(_Float16)x2.x;  bv[9]=(_Float16)x2.y;  bv[10]=(_Float16)x2.z; bv[11]=(_Float16)x2.w;
            bv[12]=(_Float16)x3.x; bv[13]=(_Float16)x3.y; bv[14]=(_Float16)x3.z; bv[15]=(_Float16)x3.w;
            c = wmma16(a, bv, c);
        }
        #pragma unroll
        for (int j = 0; j < 8; ++j) c[j] *= INV_T;
        cacc[t] = c;
    }

    // ---- row max (C layout: vgpr j, lanes 0-15 -> row j; lanes 16-31 -> row j+8)
    #pragma unroll
    for (int j = 0; j < 8; ++j) {
        float pm = cacc[0][j];
        #pragma unroll
        for (int t = 1; t < 8; ++t) pm = fmaxf(pm, cacc[t][j]);
        red[w][j + (hib << 3)][lane15] = pm;
    }
    __syncthreads();
    {
        int row = tid >> 4, ch = tid & 15;
        float m = red[0][row][ch];
        for (int w2 = 1; w2 < 8; ++w2) m = fmaxf(m, red[w2][row][ch]);
        red[0][row][ch] = m;
    }
    __syncthreads();
    if (tid < 16) {
        float m = red[0][tid][0];
        for (int ch = 1; ch < 16; ++ch) m = fmaxf(m, red[0][tid][ch]);
        rowmax[tid] = m;
    }
    __syncthreads();

    // ---- row sum of exp
    #pragma unroll
    for (int j = 0; j < 8; ++j) {
        int row = j + (hib << 3);
        float rm = rowmax[row];
        float ps = 0.f;
        #pragma unroll
        for (int t = 0; t < 8; ++t) ps += __expf(cacc[t][j] - rm);
        red[w][row][lane15] = ps;
    }
    __syncthreads();
    {
        int row = tid >> 4, ch = tid & 15;
        float ssum = 0.f;
        for (int w2 = 0; w2 < 8; ++w2) ssum += red[w2][row][ch];
        red[0][row][ch] = ssum;
    }
    __syncthreads();
    if (tid < 16) {
        float ssum = 0.f;
        for (int ch = 0; ch < 16; ++ch) ssum += red[0][tid][ch];
        rowsum[tid] = ssum;
    }
    __syncthreads();

    // ---- write attn/log_attn, scatter attn (f16) into frag as A-fragments for pass2
    #pragma unroll
    for (int t = 0; t < 8; ++t) {
        int cc = ((w << 3) + t) * 16 + lane15;         // column 0..1023
        int sstep = cc >> 5, wi = cc & 31, lo, pos;
        afrag_map(wi, lo, pos);
        #pragma unroll
        for (int j = 0; j < 8; ++j) {
            int rm = j + (hib << 3);
            float la = cacc[t][j] - rowmax[rm] - __logf(rowsum[rm]);
            float av = __expf(la);
            size_t gi = ((size_t)b * L_ + m0 + rm) * L_ + cc;
            attnOut[gi]    = av;
            logAttnOut[gi] = la;
            frag[(((sstep << 5) + rm + lo) << 4) + pos] = (_Float16)av;
        }
    }
    __syncthreads();

    // ---- pass 2: output = attn @ v. wave w owns D columns [w*64, w*64+64)
    const float* vb = v + (size_t)b * L_ * D_;
    #pragma unroll
    for (int t2 = 0; t2 < 4; ++t2) {
        v8f c = {0.f, 0.f, 0.f, 0.f, 0.f, 0.f, 0.f, 0.f};
        int dcol = (w << 6) + (t2 << 4) + lane15;
        const float* vcol = vb + dcol;
        for (int s = 0; s < 32; ++s) {
            v16h a = *(const v16h*)&frag[((s << 5) + ll) << 4];
            int k0 = (s << 5) + (hib ? 8 : 0);
            v16h bv;
            #pragma unroll
            for (int p = 0; p < 8; ++p) bv[p]     = (_Float16)vcol[(size_t)(k0 + p)      * D_];
            #pragma unroll
            for (int p = 0; p < 8; ++p) bv[8 + p] = (_Float16)vcol[(size_t)(k0 + 16 + p) * D_];
            c = wmma16(a, bv, c);
        }
        // stash into the n=0 slice of the big output (combine kernel finalizes it)
        #pragma unroll
        for (int j = 0; j < 8; ++j) {
            int rm = j + (hib << 3);
            outBig[(((size_t)b * NQ_) * L_ + m0 + rm) * D_ + dcol] = c[j];
        }
    }
}

// ---------------------------------------------------------------------------
// Kernel 2: query-gated path. One block per batch; NQ=16 is exactly one M-tile.
// ---------------------------------------------------------------------------
__global__ __launch_bounds__(256)
void qgsa_query_attn(const float* __restrict__ query, const float* __restrict__ k,
                     float* __restrict__ qattn)
{
    __shared__ __align__(32) _Float16 frag[8192];    // 16KB: query fragments (16 steps)
    __shared__ float red[8][16][16];
    __shared__ float rowmax[16];
    __shared__ float rowsum[16];

    const int tid    = threadIdx.x;
    const int w      = tid >> 5;
    const int ll     = tid & 31;
    const int lane15 = ll & 15;
    const int hib    = (ll >= 16) ? 1 : 0;
    const int b      = blockIdx.x;

    for (int e = tid; e < 16 * D_; e += 256) {
        int m = e >> 9;
        int d = e & (D_ - 1);
        float val = query[((size_t)b * NQ_ + m) * D_ + d];
        int s = d >> 5, wi = d & 31, lo, pos;
        afrag_map(wi, lo, pos);
        frag[(((s << 5) + m + lo) << 4) + pos] = (_Float16)val;
    }
    __syncthreads();

    v8f cacc[8];
    const float* kb = k + (size_t)b * L_ * D_;
    #pragma unroll
    for (int t = 0; t < 8; ++t) {
        v8f c = {0.f, 0.f, 0.f, 0.f, 0.f, 0.f, 0.f, 0.f};
        const float* krow = kb + (size_t)(((w << 3) + t) * 16 + lane15) * D_;
        for (int s = 0; s < 16; ++s) {
            v16h a = *(const v16h*)&frag[((s << 5) + ll) << 4];
            int c0 = (s << 5) + (hib ? 8 : 0);
            float4 x0 = *(const float4*)(krow + c0);
            float4 x1 = *(const float4*)(krow + c0 + 4);
            float4 x2 = *(const float4*)(krow + c0 + 16);
            float4 x3 = *(const float4*)(krow + c0 + 20);
            v16h bv;
            bv[0]=(_Float16)x0.x;  bv[1]=(_Float16)x0.y;  bv[2]=(_Float16)x0.z;  bv[3]=(_Float16)x0.w;
            bv[4]=(_Float16)x1.x;  bv[5]=(_Float16)x1.y;  bv[6]=(_Float16)x1.z;  bv[7]=(_Float16)x1.w;
            bv[8]=(_Float16)x2.x;  bv[9]=(_Float16)x2.y;  bv[10]=(_Float16)x2.z; bv[11]=(_Float16)x2.w;
            bv[12]=(_Float16)x3.x; bv[13]=(_Float16)x3.y; bv[14]=(_Float16)x3.z; bv[15]=(_Float16)x3.w;
            c = wmma16(a, bv, c);
        }
        #pragma unroll
        for (int j = 0; j < 8; ++j) c[j] *= INV_T;
        cacc[t] = c;
    }

    #pragma unroll
    for (int j = 0; j < 8; ++j) {
        float pm = cacc[0][j];
        #pragma unroll
        for (int t = 1; t < 8; ++t) pm = fmaxf(pm, cacc[t][j]);
        red[w][j + (hib << 3)][lane15] = pm;
    }
    __syncthreads();
    {
        int row = tid >> 4, ch = tid & 15;
        float m = red[0][row][ch];
        for (int w2 = 1; w2 < 8; ++w2) m = fmaxf(m, red[w2][row][ch]);
        red[0][row][ch] = m;
    }
    __syncthreads();
    if (tid < 16) {
        float m = red[0][tid][0];
        for (int ch = 1; ch < 16; ++ch) m = fmaxf(m, red[0][tid][ch]);
        rowmax[tid] = m;
    }
    __syncthreads();
    #pragma unroll
    for (int j = 0; j < 8; ++j) {
        int row = j + (hib << 3);
        float rm = rowmax[row];
        float ps = 0.f;
        #pragma unroll
        for (int t = 0; t < 8; ++t) ps += __expf(cacc[t][j] - rm);
        red[w][row][lane15] = ps;
    }
    __syncthreads();
    {
        int row = tid >> 4, ch = tid & 15;
        float ssum = 0.f;
        for (int w2 = 0; w2 < 8; ++w2) ssum += red[w2][row][ch];
        red[0][row][ch] = ssum;
    }
    __syncthreads();
    if (tid < 16) {
        float ssum = 0.f;
        for (int ch = 0; ch < 16; ++ch) ssum += red[0][tid][ch];
        rowsum[tid] = ssum;
    }
    __syncthreads();

    #pragma unroll
    for (int t = 0; t < 8; ++t) {
        int cc = ((w << 3) + t) * 16 + lane15;
        #pragma unroll
        for (int j = 0; j < 8; ++j) {
            int rm = j + (hib << 3);
            float av = __expf(cacc[t][j] - rowmax[rm]) / rowsum[rm];
            qattn[((size_t)b * NQ_ + rm) * L_ + cc] = av;
        }
    }
}

// ---------------------------------------------------------------------------
// Kernel 3: out[b,n,l,:] = output[b,l,:] + qattn[b,n,l] * v[b,l,:]
// output[b,l,:] was stashed at out[b,0,l,:]; read-before-write per thread.
// Pure streaming: 268 MB of coalesced stores -> the HBM-bound leg.
// ---------------------------------------------------------------------------
__global__ __launch_bounds__(256)
void qgsa_combine(const float* __restrict__ v, const float* __restrict__ qattn,
                  float* outBig)
{
    int b = blockIdx.x >> 10;
    int l = blockIdx.x & 1023;
    int d = threadIdx.x << 1;

    size_t stashIdx = (((size_t)b * NQ_) * L_ + l) * D_ + d;
    float2 o  = *(const float2*)&outBig[stashIdx];
    float2 vv = *(const float2*)&v[((size_t)b * L_ + l) * D_ + d];
    const float* qa = qattn + ((size_t)b * NQ_) * L_ + l;

    #pragma unroll
    for (int n = 0; n < NQ_; ++n) {
        float g = qa[(size_t)n * L_];
        float2 r;
        r.x = o.x + g * vv.x;
        r.y = o.y + g * vv.y;
        *(float2*)&outBig[(((size_t)b * NQ_ + n) * L_ + l) * D_ + d] = r;
    }
}

extern "C" void kernel_launch(void* const* d_in, const int* in_sizes, int n_in,
                              void* d_out, int out_size, void* d_ws, size_t ws_size,
                              hipStream_t stream) {
    const float* q     = (const float*)d_in[0];
    const float* k     = (const float*)d_in[1];
    const float* v     = (const float*)d_in[2];
    const float* query = (const float*)d_in[3];

    float* outBig     = (float*)d_out;                                  // [8,16,1024,512]
    float* attnOut    = outBig + (size_t)B_ * NQ_ * L_ * D_;            // [8,1024,1024]
    float* logAttnOut = attnOut + (size_t)B_ * L_ * L_;                 // [8,1024,1024]
    float* qattn      = (float*)d_ws;                                   // [8,16,1024] = 512KB

    qgsa_self_attn<<<B_ * (L_ / 16), 256, 0, stream>>>(q, k, v, outBig, attnOut, logAttnOut);
    qgsa_query_attn<<<B_, 256, 0, stream>>>(query, k, qattn);
    qgsa_combine<<<B_ * L_, 256, 0, stream>>>(v, qattn, outBig);
}
